// GroupedQueryAttention_34411277975856
// MI455X (gfx1250) — compile-verified, matrix operand order checked by reading, not run on hardware
//
#include <hip/hip_runtime.h>
#include <hip/hip_bf16.h>

#define E_   2048
#define H_   16
#define HD_  128
#define B_   4
#define N_   2048
#define BH_  (B_*H_)

typedef __bf16 bf16_t;
typedef __attribute__((ext_vector_type(16))) __bf16 bf16x16;
typedef __attribute__((ext_vector_type(8)))  __bf16 bf16x8;
typedef __attribute__((ext_vector_type(4)))  __bf16 bf16x4;
typedef __attribute__((ext_vector_type(8)))  float  f32x8;
typedef __attribute__((ext_vector_type(4)))  float  f32x4;

enum OutMode {
    OUT_PLAIN_F32    = 0,   // C[bz*strideC + m*Ncols + n] (f32)
    OUT_HEADS_F32    = 1,   // [B,H,N,HD] f32   (Q projection)
    OUT_HEADS_BF16   = 2,   // [B,H,N,HD] bf16  (K projection)
    OUT_HEADS_T_BF16 = 3,   // [B,H,HD,N] bf16  (V projection, transposed)
    OUT_ATTNOUT_F32  = 4    // [B,N,E] f32 head-concat (PV result)
};

constexpr int BM = 128, BN = 128, BK = 32;
constexpr int LDSS = 40;  // bf16 elems per LDS row: 80 B (16B-aligned, 20-bank skew)

// ---------------------------------------------------------------------------
// TN GEMM: C(M x Ncols) = scale * A(M x K, f32) * B(Ncols x K, bf16)^T + bias
// Double-buffered software pipeline:
//   A tile: global f32 -> VGPR -> cvt bf16 -> LDS (needs VALU conversion)
//   B tile: global bf16 -> LDS directly via global_load_async_to_lds_b128
//           (ASYNCcnt-tracked CDNA5 async copy, bypasses VGPRs)
// 256 threads = 8 waves; each wave computes a 32x64 sub-tile = 2x4 WMMA tiles.
// ---------------------------------------------------------------------------
template<int MODE>
__global__ __launch_bounds__(256)
void gemm_tn(const float* __restrict__ A, const bf16_t* __restrict__ Bm,
             void* __restrict__ Cout, const float* __restrict__ bias,
             float scale, int M, int Ncols, int K,
             long long strideA, long long strideB, long long strideC)
{
    __shared__ bf16_t sA[2][BM * LDSS];   // 2 x 10240 B
    __shared__ bf16_t sB[2][BN * LDSS];   // 2 x 10240 B

    const int bz = blockIdx.z;
    const float*  Ab = A  + (long long)bz * strideA;
    const bf16_t* Bb = Bm + (long long)bz * strideB;
    const int bm = blockIdx.y * BM;
    const int bn = blockIdx.x * BN;

    const int tid   = threadIdx.x;
    const int lane  = tid & 31;
    const int wave  = tid >> 5;
    const int waveM = wave >> 1;          // 0..3 -> 32 rows each
    const int waveN = wave & 1;           // 0..1 -> 64 cols each
    const int frow  = lane & 15;          // row/col within 16x16 tile
    const int fk    = (lane >> 4) << 3;   // K sub-offset: 0 or 8

    f32x8 acc[2][4] = {};
    f32x4 areg[4];

    // issue all A global loads for K-tile k0 into registers (batched, no
    // per-load waits on the critical path)
    auto load_a = [&](int k0) {
        #pragma unroll
        for (int i = 0; i < 4; ++i) {
            int idx = i * 256 + tid, row = idx >> 3, col = (idx & 7) << 2;
            areg[i] = *(const f32x4*)(Ab + (long long)(bm + row) * K + k0 + col);
        }
    };
    // convert staged A registers to bf16 and store into LDS buffer `buf`
    auto store_a = [&](int buf) {
        #pragma unroll
        for (int i = 0; i < 4; ++i) {
            int idx = i * 256 + tid, row = idx >> 3, col = (idx & 7) << 2;
            bf16x4 w;
            w[0] = (bf16_t)areg[i][0]; w[1] = (bf16_t)areg[i][1];
            w[2] = (bf16_t)areg[i][2]; w[3] = (bf16_t)areg[i][3];
            *(bf16x4*)(&sA[buf][row * LDSS + col]) = w;
        }
    };
    // async-copy the bf16 B tile straight into LDS buffer `buf`
    auto async_b = [&](int k0, int buf) {
        #pragma unroll
        for (int i = 0; i < 2; ++i) {
            int idx = i * 256 + tid, row = idx >> 2, col = (idx & 3) << 3;
            unsigned ldsoff =
                (unsigned)(uintptr_t)(&sB[buf][row * LDSS + col]);
            const bf16_t* g = Bb + (long long)(bn + row) * K + k0 + col;
            asm volatile("global_load_async_to_lds_b128 %0, %1, off"
                         :: "v"(ldsoff),
                            "v"((unsigned long long)(uintptr_t)g)
                         : "memory");
        }
    };
    auto wait_async = [&]() {
        asm volatile("s_wait_asynccnt 0x0" ::: "memory");
    };
    // fragment loads (ISA 7.12.2 16-bit layouts) + 8 WMMAs on buffer `buf`
    auto compute = [&](int buf) {
        bf16x16 af[2], bfr[4];
        #pragma unroll
        for (int tm = 0; tm < 2; ++tm) {
            const bf16_t* p = &sA[buf][(waveM * 32 + tm * 16 + frow) * LDSS + fk];
            bf16x8 lo = *(const bf16x8*)p;          // K = fk .. fk+7
            bf16x8 hi = *(const bf16x8*)(p + 16);   // K = 16+fk .. 16+fk+7
            #pragma unroll
            for (int e = 0; e < 8; ++e) { af[tm][e] = lo[e]; af[tm][8 + e] = hi[e]; }
        }
        #pragma unroll
        for (int tn = 0; tn < 4; ++tn) {
            const bf16_t* p = &sB[buf][(waveN * 64 + tn * 16 + frow) * LDSS + fk];
            bf16x8 lo = *(const bf16x8*)p;
            bf16x8 hi = *(const bf16x8*)(p + 16);
            #pragma unroll
            for (int e = 0; e < 8; ++e) { bfr[tn][e] = lo[e]; bfr[tn][8 + e] = hi[e]; }
        }
        #pragma unroll
        for (int tm = 0; tm < 2; ++tm)
            #pragma unroll
            for (int tn = 0; tn < 4; ++tn)
                acc[tm][tn] = __builtin_amdgcn_wmma_f32_16x16x32_bf16(
                    false, af[tm], false, bfr[tn],
                    (short)0, acc[tm][tn], false, false);
    };

    // ---- pipeline prologue: stage K-tile 0 into buffer 0 ----
    load_a(0);
    async_b(0, 0);
    store_a(0);
    wait_async();
    __syncthreads();

    const int nsteps = K / BK;
    int cur = 0;
    for (int s = 0; s < nsteps; ++s) {
        const bool has_next = (s + 1) < nsteps;
        if (has_next) {
            load_a((s + 1) * BK);            // global loads in flight...
            async_b((s + 1) * BK, cur ^ 1);  // ...and async DMA to other buffer
            __builtin_prefetch(Ab + (long long)(bm + (tid >> 1)) * K + (s + 2) * BK, 0, 1);
        }
        compute(cur);                        // WMMAs overlap the fetches
        if (has_next) {
            store_a(cur ^ 1);                // drain A regs -> LDS
            wait_async();                    // B tile landed in LDS
            __syncthreads();                 // single barrier per K-step
        }
        cur ^= 1;
    }

    // ---- epilogue: C/D layout: VGPR i -> M = (lane>=16 ? 8:0)+i, N = lane&15 ----
    #pragma unroll
    for (int tm = 0; tm < 2; ++tm)
        #pragma unroll
        for (int tn = 0; tn < 4; ++tn)
            #pragma unroll
            for (int i = 0; i < 8; ++i) {
                int m = bm + waveM * 32 + tm * 16 + ((lane >> 4) << 3) + i;
                int n = bn + waveN * 64 + tn * 16 + (lane & 15);
                float v = acc[tm ^ 1 ^ 1][tn][i] * scale;  // (identity; keep indices clear)
                if (bias) v += bias[n];
                if (MODE == OUT_PLAIN_F32) {
                    ((float*)Cout)[(long long)bz * strideC + (long long)m * Ncols + n] = v;
                } else if (MODE == OUT_HEADS_F32 || MODE == OUT_HEADS_BF16 ||
                           MODE == OUT_HEADS_T_BF16) {
                    int b = m >> 11, nn = m & (N_ - 1);   // m = b*N + nn
                    int h = n >> 7,  d  = n & (HD_ - 1);  // n = h*HD + d
                    if (MODE == OUT_HEADS_F32)
                        ((float*)Cout)[((long long)(b * H_ + h) * N_ + nn) * HD_ + d] = v;
                    else if (MODE == OUT_HEADS_BF16)
                        ((bf16_t*)Cout)[((long long)(b * H_ + h) * N_ + nn) * HD_ + d] = (bf16_t)v;
                    else
                        ((bf16_t*)Cout)[((long long)(b * H_ + h) * HD_ + d) * N_ + nn] = (bf16_t)v;
                } else { // OUT_ATTNOUT_F32: bz = b*H + h
                    int b = bz >> 4, h = bz & (H_ - 1);
                    ((float*)Cout)[((long long)b * N_ + m) * E_ + h * HD_ + n] = v;
                }
            }
}

// ---------------------------------------------------------------------------
// Row softmax over 2048 columns, one 256-thread block per row, in place.
// ---------------------------------------------------------------------------
__global__ __launch_bounds__(256)
void softmax_rows(float* __restrict__ attn)
{
    __shared__ float red[256];
    float* p = attn + (long long)blockIdx.x * N_;
    const int tid = threadIdx.x;

    f32x4 v0 = *(const f32x4*)(p + tid * 8);
    f32x4 v1 = *(const f32x4*)(p + tid * 8 + 4);

    float mx = -3.0e38f;
    #pragma unroll
    for (int e = 0; e < 4; ++e) { mx = fmaxf(mx, v0[e]); mx = fmaxf(mx, v1[e]); }
    red[tid] = mx; __syncthreads();
    #pragma unroll
    for (int s = 128; s > 0; s >>= 1) {
        if (tid < s) red[tid] = fmaxf(red[tid], red[tid + s]);
        __syncthreads();
    }
    mx = red[0]; __syncthreads();

    float sum = 0.0f;
    #pragma unroll
    for (int e = 0; e < 4; ++e) {
        v0[e] = __expf(v0[e] - mx); sum += v0[e];
        v1[e] = __expf(v1[e] - mx); sum += v1[e];
    }
    red[tid] = sum; __syncthreads();
    #pragma unroll
    for (int s = 128; s > 0; s >>= 1) {
        if (tid < s) red[tid] += red[tid + s];
        __syncthreads();
    }
    float inv = 1.0f / red[0];
    #pragma unroll
    for (int e = 0; e < 4; ++e) { v0[e] *= inv; v1[e] *= inv; }
    *(f32x4*)(p + tid * 8)     = v0;
    *(f32x4*)(p + tid * 8 + 4) = v1;
}

// ---------------------------------------------------------------------------
__global__ __launch_bounds__(256)
void cvt_f32_bf16(const float* __restrict__ src, bf16_t* __restrict__ dst, int n4)
{
    int i = blockIdx.x * 256 + threadIdx.x;
    if (i < n4) {
        f32x4 v = ((const f32x4*)src)[i];
        bf16x4 w;
        w[0] = (bf16_t)v[0]; w[1] = (bf16_t)v[1];
        w[2] = (bf16_t)v[2]; w[3] = (bf16_t)v[3];
        ((bf16x4*)dst)[i] = w;
    }
}

// ---------------------------------------------------------------------------
extern "C" void kernel_launch(void* const* d_in, const int* in_sizes, int n_in,
                              void* d_out, int out_size, void* d_ws, size_t ws_size,
                              hipStream_t stream)
{
    const float* q  = (const float*)d_in[0];
    const float* k  = (const float*)d_in[1];
    const float* v  = (const float*)d_in[2];
    const float* Wq = (const float*)d_in[3];
    const float* bq = (const float*)d_in[4];
    const float* Wk = (const float*)d_in[5];
    const float* bk = (const float*)d_in[6];
    const float* Wv = (const float*)d_in[7];
    const float* bv = (const float*)d_in[8];
    const float* Wo = (const float*)d_in[9];
    const float* bo = (const float*)d_in[10];

    float* out  = (float*)d_out;
    float* attn = out + (size_t)B_ * N_ * E_;           // [BH, N, N] f32

    char* ws = (char*)d_ws;
    size_t off = 0;
    bf16_t* Wq_b = (bf16_t*)(ws + off); off += (size_t)E_ * E_ * 2;
    bf16_t* Wk_b = (bf16_t*)(ws + off); off += (size_t)E_ * E_ * 2;
    bf16_t* Wv_b = (bf16_t*)(ws + off); off += (size_t)E_ * E_ * 2;
    bf16_t* Wo_b = (bf16_t*)(ws + off); off += (size_t)E_ * E_ * 2;
    float*  Qh   = (float*) (ws + off); off += (size_t)B_ * N_ * E_ * 4;  // [BH,N,HD] f32
    bf16_t* Kh   = (bf16_t*)(ws + off); off += (size_t)B_ * N_ * E_ * 2;  // [BH,N,HD] bf16
    bf16_t* Vt   = (bf16_t*)(ws + off); off += (size_t)B_ * N_ * E_ * 2;  // [BH,HD,N] bf16
    float*  Oc   = (float*) (ws + off); off += (size_t)B_ * N_ * E_ * 4;  // [B,N,E]  f32

    dim3 blk(256);

    // weight conversion
    int n4 = E_ * E_ / 4;
    cvt_f32_bf16<<<(n4 + 255) / 256, blk, 0, stream>>>(Wq, Wq_b, n4);
    cvt_f32_bf16<<<(n4 + 255) / 256, blk, 0, stream>>>(Wk, Wk_b, n4);
    cvt_f32_bf16<<<(n4 + 255) / 256, blk, 0, stream>>>(Wv, Wv_b, n4);
    cvt_f32_bf16<<<(n4 + 255) / 256, blk, 0, stream>>>(Wo, Wo_b, n4);

    dim3 gproj(E_ / BN, (B_ * N_) / BM, 1);
    // Q/K/V projections
    gemm_tn<OUT_HEADS_F32>   <<<gproj, blk, 0, stream>>>(q, Wq_b, Qh, bq, 1.0f,
        B_ * N_, E_, E_, 0, 0, 0);
    gemm_tn<OUT_HEADS_BF16>  <<<gproj, blk, 0, stream>>>(k, Wk_b, Kh, bk, 1.0f,
        B_ * N_, E_, E_, 0, 0, 0);
    gemm_tn<OUT_HEADS_T_BF16><<<gproj, blk, 0, stream>>>(v, Wv_b, Vt, bv, 1.0f,
        B_ * N_, E_, E_, 0, 0, 0);

    // scores = Qh @ Kh^T / sqrt(HD)   (batched over BH)
    gemm_tn<OUT_PLAIN_F32><<<dim3(N_ / BN, N_ / BM, BH_), blk, 0, stream>>>(
        Qh, Kh, attn, nullptr, 0.08838834764831845f,
        N_, N_, HD_, (long long)N_ * HD_, (long long)N_ * HD_, (long long)N_ * N_);

    // softmax in place (also the attn_weights output)
    softmax_rows<<<BH_ * N_, blk, 0, stream>>>(attn);

    // attn_out = P @ V   (A = f32 attn probs, B = transposed bf16 V)
    gemm_tn<OUT_ATTNOUT_F32><<<dim3(HD_ / BN, N_ / BM, BH_), blk, 0, stream>>>(
        attn, Vt, Oc, nullptr, 1.0f,
        N_, HD_, N_, (long long)N_ * N_, (long long)HD_ * N_, 0);

    // out = attn_out @ Wo^T + bo
    gemm_tn<OUT_PLAIN_F32><<<gproj, blk, 0, stream>>>(
        Oc, Wo_b, out, bo, 1.0f, B_ * N_, E_, E_, 0, 0, 0);
}
